// Net_52020643889507
// MI455X (gfx1250) — compile-verified, matrix operand order checked by reading, not run on hardware
//
#include <hip/hip_runtime.h>
#include <hip/hip_bf16.h>

// ---------------------------------------------------------------------------
// GraphConv GNN for MI455X (gfx1250, wave32).
//   Layer: agg[dst] += h[src]*ew  (scatter, native L2 f32 atomics)
//          h' = relu(agg @ W_rel^T + b + h @ W_root^T)   (f32 WMMA 16x16x4)
//   Then global_add_pool + 2-layer MLP head.
// All live data (<80MB) fits in 192MB L2 -> scatter/gather is L2-resident;
// the workload is bound by edge gather + f32 atomic throughput, not FLOPs,
// so the GEMM runs in exact f32 on the WMMA path.
// ---------------------------------------------------------------------------

typedef __attribute__((ext_vector_type(2)))  float    v2f;
typedef __attribute__((ext_vector_type(8)))  float    v8f;
typedef __attribute__((ext_vector_type(16))) _Float16 v16h;

#define DIM 128

// Native fire-and-forget f32 atomic add (global_atomic_add_u32/f32, no CAS loop).
__device__ __forceinline__ void atomic_add_f32(float* p, float v) {
#if defined(__HIP_DEVICE_COMPILE__)
    unsafeAtomicAdd(p, v);           // lowers to global_atomic_add_f32
#else
    __hip_atomic_fetch_add(p, v, __ATOMIC_RELAXED, __HIP_MEMORY_SCOPE_AGENT);
#endif
}

// -------------------------- edge scatter ------------------------------------
// One wave per edge: lane i handles floats [4i, 4i+4) of the 128-dim feature.
__global__ void scatter_kernel(const float* __restrict__ h,
                               const int*   __restrict__ src,
                               const int*   __restrict__ dst,
                               const float* __restrict__ ew,
                               float*       __restrict__ agg,
                               int num_edges) {
    int wave = (blockIdx.x * blockDim.x + threadIdx.x) >> 5;
    int lane = threadIdx.x & 31;
    if (wave >= num_edges) return;
    int   s = src[wave];
    int   d = dst[wave];
    float w = ew[wave];
    const float4* hs = (const float4*)(h + (size_t)s * DIM);
    float4 v = hs[lane];                       // global_load_b128, coalesced
    float* ad = agg + (size_t)d * DIM + lane * 4;
    atomic_add_f32(ad + 0, v.x * w);
    atomic_add_f32(ad + 1, v.y * w);
    atomic_add_f32(ad + 2, v.z * w);
    atomic_add_f32(ad + 3, v.w * w);
}

// -------------------------- dual GEMM + bias + ReLU -------------------------
// out = relu(A1 @ W1^T + A2 @ W2^T + bias), all [*,128] row-major, W [128,128].
// One wave computes one 16x16 output tile; both GEMMs accumulate into one v8f.
__global__ void gemm_dual_relu(const float* __restrict__ A1,
                               const float* __restrict__ W1,
                               const float* __restrict__ A2,
                               const float* __restrict__ W2,
                               const float* __restrict__ bias,
                               float*       __restrict__ out,
                               int n_rows) {
    const int ntiles_n = DIM / 16;                       // 8
    int wave = (blockIdx.x * blockDim.x + threadIdx.x) >> 5;
    int lane = threadIdx.x & 31;
    int mt = wave / ntiles_n;
    int nt = wave % ntiles_n;
    int m0 = mt * 16;
    if (m0 >= n_rows) return;                            // wave-uniform branch
    int n0 = nt * 16;
    int hi = lane >> 4;                                  // half-wave select
    int l  = lane & 15;

    const float* arow1 = A1 + (size_t)(m0 + l) * DIM;
    const float* wrow1 = W1 + (size_t)(n0 + l) * DIM;
    const float* arow2 = A2 + (size_t)(m0 + l) * DIM;
    const float* wrow2 = W2 + (size_t)(n0 + l) * DIM;

    v8f acc = {};

#if __has_builtin(__builtin_amdgcn_wmma_f32_16x16x4_f32)
    // ISA 16x4 f32 A-layout: lanes 0-15 hold {K,K+1}, lanes 16-31 hold {K+2,K+3}.
    // B (4x16) mirrors this per column; B[k][n] = W[n][k].
    #pragma unroll
    for (int k = 0; k < DIM; k += 4) {
        int kk = k + 2 * hi;
        v2f a, b;
        a.x = arow1[kk];  a.y = arow1[kk + 1];
        b.x = wrow1[kk];  b.y = wrow1[kk + 1];
        acc = __builtin_amdgcn_wmma_f32_16x16x4_f32(false, a, false, b,
                                                    (short)0, acc, false, false);
    }
    #pragma unroll
    for (int k = 0; k < DIM; k += 4) {
        int kk = k + 2 * hi;
        v2f a, b;
        a.x = arow2[kk];  a.y = arow2[kk + 1];
        b.x = wrow2[kk];  b.y = wrow2[kk + 1];
        acc = __builtin_amdgcn_wmma_f32_16x16x4_f32(false, a, false, b,
                                                    (short)0, acc, false, false);
    }
#else
    // Fallback: codegen-confirmed f16 WMMA (16x16x32), f32 accumulate.
    #pragma unroll
    for (int kb = 0; kb < DIM / 32; ++kb) {
        v16h a1, b1, a2, b2;
        #pragma unroll
        for (int i = 0; i < 16; ++i) {
            int v = i >> 1, r = i & 1;
            int klocal = (v < 4) ? (2 * v + r + 8 * hi)
                                 : (16 + 2 * (v - 4) + r + 8 * hi);
            int k = kb * 32 + klocal;
            a1[i] = (_Float16)arow1[k];
            b1[i] = (_Float16)wrow1[k];
            a2[i] = (_Float16)arow2[k];
            b2[i] = (_Float16)wrow2[k];
        }
        acc = __builtin_amdgcn_wmma_f32_16x16x32_f16(false, a1, false, b1,
                                                     (short)0, acc, false, false);
        acc = __builtin_amdgcn_wmma_f32_16x16x32_f16(false, a2, false, b2,
                                                     (short)0, acc, false, false);
    }
#endif

    float bn = bias[n0 + l];
    // C/D layout: VGPR r -> row m0+r (lanes 0-15) / m0+r+8 (lanes 16-31), col n0+l.
    #pragma unroll
    for (int r = 0; r < 8; ++r) {
        int row = m0 + r + hi * 8;
        float v = acc[r] + bn;
        out[(size_t)row * DIM + n0 + l] = v > 0.0f ? v : 0.0f;
    }
}

// -------------------------- per-graph pooling -------------------------------
__global__ void pool_kernel(const float* __restrict__ h,
                            const int*   __restrict__ batch,
                            float*       __restrict__ pooled,
                            int n_nodes) {
    int wave = (blockIdx.x * blockDim.x + threadIdx.x) >> 5;
    int lane = threadIdx.x & 31;
    if (wave >= n_nodes) return;
    int g = batch[wave];
    const float4* hn = (const float4*)(h + (size_t)wave * DIM);
    float4 v = hn[lane];
    float* pg = pooled + (size_t)g * DIM + lane * 4;
    atomic_add_f32(pg + 0, v.x);
    atomic_add_f32(pg + 1, v.y);
    atomic_add_f32(pg + 2, v.z);
    atomic_add_f32(pg + 3, v.w);
}

// -------------------------- MLP head ----------------------------------------
// One 128-thread block per graph: fc1 (128x128) + relu, then fc2 (1x128).
__global__ void head_kernel(const float* __restrict__ pooled,
                            const float* __restrict__ Wfc1,
                            const float* __restrict__ bfc1,
                            const float* __restrict__ Wfc2,
                            const float* __restrict__ bfc2,
                            float*       __restrict__ out) {
    __shared__ float s[DIM];
    int g = blockIdx.x;
    int d = threadIdx.x;
    const float* p = pooled + (size_t)g * DIM;
    const float* w = Wfc1 + (size_t)d * DIM;
    float acc = bfc1[d];
    #pragma unroll 4
    for (int k = 0; k < DIM; ++k) acc += w[k] * p[k];
    float f = acc > 0.0f ? acc : 0.0f;
    s[d] = f * Wfc2[d];
    __syncthreads();
    #pragma unroll
    for (int off = 64; off > 0; off >>= 1) {
        if (d < off) s[d] += s[d + off];
        __syncthreads();
    }
    if (d == 0) out[g] = s[0] + bfc2[0];
}

// ---------------------------------------------------------------------------
extern "C" void kernel_launch(void* const* d_in, const int* in_sizes, int n_in,
                              void* d_out, int out_size, void* d_ws, size_t ws_size,
                              hipStream_t stream) {
    const float* x        = (const float*)d_in[0];
    const int*   edge_idx = (const int*)  d_in[1];
    const int*   batch    = (const int*)  d_in[2];
    const float* ew       = (const float*)d_in[3];
    const float* W_rel1   = (const float*)d_in[4];
    const float* b_rel1   = (const float*)d_in[5];
    const float* W_root1  = (const float*)d_in[6];
    const float* W_rel_s  = (const float*)d_in[7];
    const float* b_rel_s  = (const float*)d_in[8];
    const float* W_root_s = (const float*)d_in[9];
    const float* W_fc1    = (const float*)d_in[10];
    const float* b_fc1    = (const float*)d_in[11];
    const float* W_fc2    = (const float*)d_in[12];
    const float* b_fc2    = (const float*)d_in[13];
    float*       out      = (float*)d_out;

    const int N = in_sizes[0] / DIM;      // 50000
    const int E = in_sizes[1] / 2;        // 800000
    const int G = out_size;               // 512

    float* ws     = (float*)d_ws;
    float* h_a    = ws;
    float* h_b    = ws + (size_t)N * DIM;
    float* agg    = ws + (size_t)2 * N * DIM;
    float* pooled = ws + (size_t)3 * N * DIM;

    const int* src = edge_idx;
    const int* dst = edge_idx + E;

    const int scatter_blocks = (E * 32 + 255) / 256;
    const int gemm_tiles     = ((N + 15) / 16) * (DIM / 16);
    const int gemm_blocks    = (gemm_tiles * 32 + 255) / 256;
    const int pool_blocks    = (N * 32 + 255) / 256;

    const float* h_in  = x;
    float*       h_out = h_a;
    for (int layer = 0; layer < 3; ++layer) {
        const float* Wr = (layer == 0) ? W_rel1  : W_rel_s  + (size_t)(layer - 1) * DIM * DIM;
        const float* br = (layer == 0) ? b_rel1  : b_rel_s  + (size_t)(layer - 1) * DIM;
        const float* Wo = (layer == 0) ? W_root1 : W_root_s + (size_t)(layer - 1) * DIM * DIM;

        hipMemsetAsync(agg, 0, (size_t)N * DIM * sizeof(float), stream);
        scatter_kernel<<<scatter_blocks, 256, 0, stream>>>(h_in, src, dst, ew, agg, E);
        gemm_dual_relu<<<gemm_blocks, 256, 0, stream>>>(agg, Wr, h_in, Wo, br, h_out, N);

        h_in  = h_out;
        h_out = (h_out == h_a) ? h_b : h_a;
    }

    hipMemsetAsync(pooled, 0, (size_t)G * DIM * sizeof(float), stream);
    pool_kernel<<<pool_blocks, 256, 0, stream>>>(h_in, batch, pooled, N);
    head_kernel<<<G, DIM, 0, stream>>>(pooled, W_fc1, b_fc1, W_fc2, b_fc2, out);
}
